// SemiringLayer_45097156608126
// MI455X (gfx1250) — compile-verified, matrix operand order checked by reading, not run on hardware
//
#include <hip/hip_runtime.h>
#include <hip/hip_bf16.h>
#include <math.h>

typedef __attribute__((ext_vector_type(16))) _Float16 v16h;
typedef __attribute__((ext_vector_type(8)))  _Float16 v8h;
typedef __attribute__((ext_vector_type(4)))  _Float16 v4h;
typedef __attribute__((ext_vector_type(8)))  float    v8f;

#define TILE_BM 128
#define TILE_BN 128
#define TILE_BK 32
#define LDT     40   // padded LDS row stride in halfs (32 + 8)

// ---- CDNA5 async global->LDS copy (per-lane 16B), tracked by ASYNCcnt ----
__device__ __forceinline__ void async_copy_b128(unsigned lds_off, const _Float16* gptr) {
    asm volatile("global_load_async_to_lds_b128 %0, %1, off"
                 :: "v"(lds_off), "v"(gptr)
                 : "memory");
}
__device__ __forceinline__ void wait_asynccnt0() {
    asm volatile("s_wait_asynccnt 0x0" ::: "memory");
}
__device__ __forceinline__ unsigned lds_addr_u32(const void* p) {
    // generic (flat) LDS pointer: low 32 bits are the wave-relative LDS offset
    return (unsigned)(unsigned long long)(uintptr_t)p;
}

// ---------------- Stage 1: per-block max reduction ----------------
__global__ __launch_bounds__(256) void reduce_max_stage1(const float* __restrict__ x,
                                                         long long n,
                                                         float* __restrict__ partials) {
    __shared__ float sm[256];
    float m = -INFINITY;
    for (long long i = (long long)blockIdx.x * 256 + threadIdx.x; i < n;
         i += (long long)gridDim.x * 256) {
        m = fmaxf(m, x[i]);
    }
    sm[threadIdx.x] = m;
    __syncthreads();
    for (int s = 128; s > 0; s >>= 1) {
        if (threadIdx.x < (unsigned)s) sm[threadIdx.x] = fmaxf(sm[threadIdx.x], sm[threadIdx.x + s]);
        __syncthreads();
    }
    if (threadIdx.x == 0) partials[blockIdx.x] = sm[0];
}

// ---------------- Stage 2: finalize both maxes in one block ----------------
__global__ __launch_bounds__(256) void reduce_max_stage2(const float* __restrict__ p1, int n1,
                                                         const float* __restrict__ p2, int n2,
                                                         float* __restrict__ maxes) {
    __shared__ float sm[256];
    float m = -INFINITY;
    for (int i = threadIdx.x; i < n1; i += 256) m = fmaxf(m, p1[i]);
    sm[threadIdx.x] = m;
    __syncthreads();
    for (int s = 128; s > 0; s >>= 1) {
        if (threadIdx.x < (unsigned)s) sm[threadIdx.x] = fmaxf(sm[threadIdx.x], sm[threadIdx.x + s]);
        __syncthreads();
    }
    if (threadIdx.x == 0) maxes[0] = sm[0];
    __syncthreads();
    m = -INFINITY;
    for (int i = threadIdx.x; i < n2; i += 256) m = fmaxf(m, p2[i]);
    sm[threadIdx.x] = m;
    __syncthreads();
    for (int s = 128; s > 0; s >>= 1) {
        if (threadIdx.x < (unsigned)s) sm[threadIdx.x] = fmaxf(sm[threadIdx.x], sm[threadIdx.x + s]);
        __syncthreads();
    }
    if (threadIdx.x == 0) maxes[1] = sm[0];
}

// ---------------- exp(x - max) -> f16, b128 load / b64 store ----------------
__global__ __launch_bounds__(256) void exp_cvt_f16(const float* __restrict__ x,
                                                   const float* __restrict__ mx,
                                                   _Float16* __restrict__ y,
                                                   long long n4) {
    long long i = ((long long)blockIdx.x * 256 + threadIdx.x);
    if (i >= n4) return;
    long long base = i * 4;
    float m = mx[0];
    float4 v = *(const float4*)&x[base];
    v4h o;
    o[0] = (_Float16)__expf(v.x - m);
    o[1] = (_Float16)__expf(v.y - m);
    o[2] = (_Float16)__expf(v.z - m);
    o[3] = (_Float16)__expf(v.w - m);
    *(v4h*)&y[base] = o;
}

// ---------------- exp(w - max_w) + transpose via LDS tiles ----------------
// w[D][O] fp32 -> wT[O][D] f16; 64x64 tile per block, coalesced both sides
__global__ __launch_bounds__(256) void exp_cvt_wT(const float* __restrict__ w,
                                                  const float* __restrict__ mx,
                                                  _Float16* __restrict__ wT,
                                                  int D, int O) {
    __shared__ _Float16 t[64][65];
    const int o0 = blockIdx.x * 64;
    const int d0 = blockIdx.y * 64;
    const float m = mx[0];
    const int col   = threadIdx.x & 63;   // fast axis
    const int rbase = threadIdx.x >> 6;   // 0..3
#pragma unroll
    for (int r = 0; r < 16; ++r) {
        int d = rbase * 16 + r;           // 0..63
        t[d][col] = (_Float16)__expf(w[(size_t)(d0 + d) * O + (o0 + col)] - m);
    }
    __syncthreads();
#pragma unroll
    for (int r = 0; r < 16; ++r) {
        int o = rbase * 16 + r;           // 0..63
        wT[(size_t)(o0 + o) * D + (d0 + col)] = t[col][o];
    }
}

// ---------------- WMMA GEMM + log epilogue ----------------
// A:  [M][K] f16 (exp(inputs - max_in)), BT: [N][K] f16 (exp(w - max_w)^T)
// out[m][n] = log(sum_k A[m][k]*BT[n][k]) + (max_in + max_w) + bias[n]
// Double-buffered LDS, async global->LDS copies overlap WMMA compute.
__global__ __launch_bounds__(256) void gemm_lse_wmma(const _Float16* __restrict__ A,
                                                     const _Float16* __restrict__ BT,
                                                     const float* __restrict__ bias,
                                                     const float* __restrict__ maxes,
                                                     float* __restrict__ out,
                                                     int M, int N, int K) {
    __shared__ _Float16 lA[2][TILE_BM * LDT];
    __shared__ _Float16 lB[2][TILE_BN * LDT];

    const int tid  = threadIdx.x;
    const int lane = tid & 31;
    const int wave = tid >> 5;        // 0..7
    const int wm   = wave >> 2;       // 0..1  -> 64 rows each
    const int wn   = wave & 3;        // 0..3  -> 32 cols each
    const int half = lane >> 4;       // 0/1 (K-half select per 16-bit A layout)
    const int l15  = lane & 15;

    const int bm = blockIdx.y * TILE_BM;
    const int bn = blockIdx.x * TILE_BN;

    // this thread's two copy chunks (8 halfs each) per operand per tile
    const int c0_row = tid >> 2;               // chunk 0: c = tid
    const int c0_kc  = (tid & 3) << 3;
    const int c1_row = (tid + 256) >> 2;       // chunk 1: c = tid + 256
    const int c1_kc  = ((tid + 256) & 3) << 3;

    auto issue_tile = [&](int k0, int buf) {
        async_copy_b128(lds_addr_u32(&lA[buf][c0_row * LDT + c0_kc]),
                        &A[(size_t)(bm + c0_row) * K + (k0 + c0_kc)]);
        async_copy_b128(lds_addr_u32(&lB[buf][c0_row * LDT + c0_kc]),
                        &BT[(size_t)(bn + c0_row) * K + (k0 + c0_kc)]);
        async_copy_b128(lds_addr_u32(&lA[buf][c1_row * LDT + c1_kc]),
                        &A[(size_t)(bm + c1_row) * K + (k0 + c1_kc)]);
        async_copy_b128(lds_addr_u32(&lB[buf][c1_row * LDT + c1_kc]),
                        &BT[(size_t)(bn + c1_row) * K + (k0 + c1_kc)]);
    };

    v8f acc[4][2];
#pragma unroll
    for (int mt = 0; mt < 4; ++mt)
#pragma unroll
        for (int nt = 0; nt < 2; ++nt)
            acc[mt][nt] = (v8f){0.f, 0.f, 0.f, 0.f, 0.f, 0.f, 0.f, 0.f};

    const int steps = K / TILE_BK;
    issue_tile(0, 0);                 // prologue: tile 0 in flight

    for (int ks = 0; ks < steps; ++ks) {
        const int buf = ks & 1;
        wait_asynccnt0();             // own tile-ks data landed in LDS
        __syncthreads();              // all waves' data landed; all done reading buf^1

        if (ks + 1 < steps)           // overlap next tile's copy with this tile's WMMAs
            issue_tile((ks + 1) * TILE_BK, buf ^ 1);

        // ---- fragments (16-bit A layout: lanes 0-15 K=0..7/16..23, lanes 16-31 K=8..15/24..31)
        v16h af[4];
#pragma unroll
        for (int mt = 0; mt < 4; ++mt) {
            const _Float16* p = &lA[buf][(wm * 64 + mt * 16 + l15) * LDT + half * 8];
            v8h lo = *(const v8h*)p;
            v8h hi = *(const v8h*)(p + 16);
            af[mt] = __builtin_shufflevector(lo, hi, 0, 1, 2, 3, 4, 5, 6, 7,
                                             8, 9, 10, 11, 12, 13, 14, 15);
        }
        v16h bf[2];
#pragma unroll
        for (int nt = 0; nt < 2; ++nt) {
            const _Float16* p = &lB[buf][(wn * 32 + nt * 16 + l15) * LDT + half * 8];
            v8h lo = *(const v8h*)p;
            v8h hi = *(const v8h*)(p + 16);
            bf[nt] = __builtin_shufflevector(lo, hi, 0, 1, 2, 3, 4, 5, 6, 7,
                                             8, 9, 10, 11, 12, 13, 14, 15);
        }

        // ---- 8 WMMAs per K-step ----
#pragma unroll
        for (int mt = 0; mt < 4; ++mt)
#pragma unroll
            for (int nt = 0; nt < 2; ++nt)
                acc[mt][nt] = __builtin_amdgcn_wmma_f32_16x16x32_f16(
                    false, af[mt], false, bf[nt], (short)0, acc[mt][nt], false, false);
    }

    // ---- epilogue: log + maxes + bias ----
    const float msum = maxes[0] + maxes[1];
#pragma unroll
    for (int mt = 0; mt < 4; ++mt) {
#pragma unroll
        for (int nt = 0; nt < 2; ++nt) {
            const int n = bn + wn * 32 + nt * 16 + l15;
            const float bv = bias[n];
#pragma unroll
            for (int v = 0; v < 8; ++v) {
                const int m = bm + wm * 64 + mt * 16 + half * 8 + v;
                out[(size_t)m * N + n] = __logf(acc[mt][nt][v]) + msum + bv;
            }
        }
    }
}

extern "C" void kernel_launch(void* const* d_in, const int* in_sizes, int n_in,
                              void* d_out, int out_size, void* d_ws, size_t ws_size,
                              hipStream_t stream) {
    const float* x    = (const float*)d_in[0];   // [B,S,D] fp32
    const float* w    = (const float*)d_in[1];   // [D,O]   fp32
    const float* bias = (const float*)d_in[2];   // [O]     fp32
    float* out = (float*)d_out;

    const long long nX = (long long)in_sizes[0]; // B*S*D
    const long long nW = (long long)in_sizes[1]; // D*O
    const int O = in_sizes[2];
    const int D = (int)(nW / O);
    const int M = (int)(nX / D);

    // ---- workspace layout ----
    char* ws = (char*)d_ws;
    float* maxes  = (float*)ws;                          // [2]
    float* partIn = (float*)(ws + 256);                  // [4096]
    const int PB_IN = 4096;
    const int PB_W  = 512;
    float* partW  = (float*)(ws + 256 + PB_IN * 4);      // [512]
    _Float16* hIn = (_Float16*)(ws + 32768);             // [M][D] f16
    _Float16* hWT = (_Float16*)(ws + 32768 + (size_t)M * D * 2); // [O][D] f16

    // 1) global maxes (two-stage, deterministic)
    reduce_max_stage1<<<PB_IN, 256, 0, stream>>>(x, nX, partIn);
    reduce_max_stage1<<<PB_W, 256, 0, stream>>>(w, nW, partW);
    reduce_max_stage2<<<1, 256, 0, stream>>>(partIn, PB_IN, partW, PB_W, maxes);

    // 2) exp(inputs - max_in) -> f16 [M][D]
    {
        long long n4 = nX / 4;
        int blocks = (int)((n4 + 255) / 256);
        exp_cvt_f16<<<blocks, 256, 0, stream>>>(x, maxes + 0, hIn, n4);
    }
    // 3) exp(w - max_w) -> f16 transposed [O][D], tiled
    {
        dim3 grid(O / 64, D / 64);
        exp_cvt_wT<<<grid, 256, 0, stream>>>(w, maxes + 1, hWT, D, O);
    }
    // 4) WMMA GEMM + log epilogue
    dim3 grid(O / TILE_BN, M / TILE_BM);
    gemm_lse_wmma<<<grid, 256, 0, stream>>>(hIn, hWT, bias, maxes, out, M, O, D);
}
